// Mulithead_Attention_14611478741492
// MI455X (gfx1250) — matrix operand order repaired and checked by reading, not compiled, not run on hardware
//
#include <hip/hip_runtime.h>

// ---------- problem constants ----------
#define D_MODEL 1024
#define D_HEAD  64
#define HEADS   16
#define BATCH   2
#define SEQ     2048

typedef _Float16 v16h __attribute__((ext_vector_type(16)));
typedef _Float16 v8h  __attribute__((ext_vector_type(8)));
typedef float    v8f  __attribute__((ext_vector_type(8)));

__device__ __forceinline__ v8f wmma_f16(v16h a, v16h b, v8f c) {
    // D = A(16x32 f16) * B(32x16 f16) + C(16x16 f32)
    return __builtin_amdgcn_wmma_f32_16x16x32_f16(
        /*neg_a=*/false, a, /*neg_b=*/false, b,
        /*c_mod=*/(short)0, c, /*reuse_a=*/false, /*reuse_b=*/false);
}

// Load a 16x32 f16 A-fragment (row-major source, row = lane%16):
//   elements 0..7  <- K = 8*laneHalf + 0..7
//   elements 8..15 <- K = 8*laneHalf + 16..23
__device__ __forceinline__ v16h load_afrag(const _Float16* rowptr, int lh) {
    v16h a;
    *((v8h*)&a)     = *(const v8h*)(rowptr + 8 * lh);
    *((v8h*)&a + 1) = *(const v8h*)(rowptr + 8 * lh + 16);
    return a;
}

// Load a 32x16 f16 B-fragment (source row-major over K, col = lane%16):
//   elements 0..7  <- K = 8*laneHalf + 0..7
//   elements 8..15 <- K = 8*laneHalf + 16..23
__device__ __forceinline__ v16h load_bfrag(const _Float16* colptr, int lh) {
    v16h b;
    *((v8h*)&b)     = *(const v8h*)(colptr + 8 * lh);
    *((v8h*)&b + 1) = *(const v8h*)(colptr + 8 * lh + 16);
    return b;
}

// ---------- conversion kernels ----------
__global__ void cvt_f32_f16(const float* __restrict__ in, _Float16* __restrict__ out, int n) {
    int i = blockIdx.x * blockDim.x + threadIdx.x;
    if (i < n) out[i] = (_Float16)in[i];
}

// in [H][M][N] fp32  ->  out [H][N][M] f16  (weight transpose so K is contiguous)
__global__ void cvt_transpose_w(const float* __restrict__ in, _Float16* __restrict__ out,
                                int H, int M, int N) {
    int i = blockIdx.x * blockDim.x + threadIdx.x;
    int total = H * M * N;
    if (i >= total) return;
    int n = i % N;
    int m = (i / N) % M;
    int h = i / (N * M);
    out[((long)h * N + n) * M + m] = (_Float16)in[i];
}

// ---------- QKV projection ----------
// 64x32 output tile per wave (4Mx2N register blocking: 8 WMMAs per 12 b128 loads).
// grid: (16, B*H, 3)  block: 128 (4 waves).
// which==0 -> Q[b,h,S,64] row-major, 1 -> K row-major, 2 -> V transposed [b,h,64,S].
__global__ void qkv_proj(const _Float16* __restrict__ xh,
                         const _Float16* __restrict__ wq_t,
                         const _Float16* __restrict__ wk_t,
                         const _Float16* __restrict__ wv_t,
                         _Float16* __restrict__ Qh,
                         _Float16* __restrict__ Kh,
                         _Float16* __restrict__ Vt) {
    const int lane = threadIdx.x & 31;
    const int wv   = threadIdx.x >> 5;
    const int lh   = lane >> 4;   // lane half: 0/1
    const int l16  = lane & 15;
    const int bh   = blockIdx.y;
    const int b    = bh / HEADS;
    const int h    = bh % HEADS;
    const int which = blockIdx.z;

    // 64 wave-tiles per (b,h,which): 32 M-blocks (64 rows) x 2 N-blocks (32 cols)
    const int tile = blockIdx.x * 4 + wv;
    const int m0   = (tile >> 1) * 64;
    const int n0   = (tile & 1) * 32;

    const _Float16* w   = (which == 0) ? wq_t : (which == 1) ? wk_t : wv_t;
    const _Float16* wr0 = w + ((long)h * D_HEAD + n0 + l16) * D_MODEL;  // cols n0..n0+15
    const _Float16* wr1 = wr0 + (long)16 * D_MODEL;                     // cols n0+16..+31
    const _Float16* xr0 = xh + ((long)b * SEQ + m0 + l16) * D_MODEL;    // rows m0..+15
    const _Float16* xr1 = xr0 + (long)16 * D_MODEL;
    const _Float16* xr2 = xr0 + (long)32 * D_MODEL;
    const _Float16* xr3 = xr0 + (long)48 * D_MODEL;

    v8f c00 = {}, c01 = {}, c10 = {}, c11 = {};
    v8f c20 = {}, c21 = {}, c30 = {}, c31 = {};
    for (int k0 = 0; k0 < D_MODEL; k0 += 32) {
        __builtin_prefetch(xr0 + k0 + 32, 0, 3);
        __builtin_prefetch(wr0 + k0 + 32, 0, 3);
        v16h b0 = load_bfrag(wr0 + k0, lh);
        v16h b1 = load_bfrag(wr1 + k0, lh);
        v16h a0 = load_afrag(xr0 + k0, lh);
        v16h a1 = load_afrag(xr1 + k0, lh);
        v16h a2 = load_afrag(xr2 + k0, lh);
        v16h a3 = load_afrag(xr3 + k0, lh);
        c00 = wmma_f16(a0, b0, c00);  c01 = wmma_f16(a0, b1, c01);
        c10 = wmma_f16(a1, b0, c10);  c11 = wmma_f16(a1, b1, c11);
        c20 = wmma_f16(a2, b0, c20);  c21 = wmma_f16(a2, b1, c21);
        c30 = wmma_f16(a3, b0, c30);  c31 = wmma_f16(a3, b1, c31);
    }

    if (which == 2) {
        // V stored transposed: [b,h, d(=64), S]
        _Float16* oc0 = Vt + ((long)bh * D_HEAD + n0 + l16) * SEQ;
        _Float16* oc1 = oc0 + (long)16 * SEQ;
        for (int j = 0; j < 8; ++j) {
            int r = m0 + j + 8 * lh;
            oc0[r]      = (_Float16)c00[j];  oc1[r]      = (_Float16)c01[j];
            oc0[r + 16] = (_Float16)c10[j];  oc1[r + 16] = (_Float16)c11[j];
            oc0[r + 32] = (_Float16)c20[j];  oc1[r + 32] = (_Float16)c21[j];
            oc0[r + 48] = (_Float16)c30[j];  oc1[r + 48] = (_Float16)c31[j];
        }
    } else {
        _Float16* o = ((which == 0) ? Qh : Kh) + (long)bh * SEQ * D_HEAD;
        for (int j = 0; j < 8; ++j) {
            long r0 = (long)(m0 + j + 8 * lh) * D_HEAD + n0 + l16;
            o[r0]                        = (_Float16)c00[j];
            o[r0 + 16]                   = (_Float16)c01[j];
            o[r0 + (long)16 * D_HEAD]      = (_Float16)c10[j];
            o[r0 + (long)16 * D_HEAD + 16] = (_Float16)c11[j];
            o[r0 + (long)32 * D_HEAD]      = (_Float16)c20[j];
            o[r0 + (long)32 * D_HEAD + 16] = (_Float16)c21[j];
            o[r0 + (long)48 * D_HEAD]      = (_Float16)c30[j];
            o[r0 + (long)48 * D_HEAD + 16] = (_Float16)c31[j];
        }
    }
}

// ---------- flash attention core ----------
// grid: (S/64, B*H)  block: 128 (4 waves).  One wave = one 16-row query tile.
__global__ void attn_kernel(const _Float16* __restrict__ Qh,
                            const _Float16* __restrict__ Kh,
                            const _Float16* __restrict__ Vt,
                            _Float16* __restrict__ ctx) {
    __shared__ __align__(32) _Float16 plds_all[4][16 * 32];

    const int lane = threadIdx.x & 31;
    const int wv   = threadIdx.x >> 5;
    const int lh   = lane >> 4;
    const int l16  = lane & 15;
    const int bh   = blockIdx.y;
    const int b    = bh / HEADS;
    const int h    = bh % HEADS;
    const int m0   = (blockIdx.x * 4 + wv) * 16;

    // resident Q fragments (16 x 64 split in two K-chunks of 32)
    const _Float16* qr = Qh + ((long)bh * SEQ + m0 + l16) * D_HEAD;
    v16h qf0 = load_afrag(qr, lh);
    v16h qf1 = load_afrag(qr + 32, lh);

    const _Float16* kbase = Kh + (long)bh * SEQ * D_HEAD;
    const _Float16* vbase = Vt + (long)bh * D_HEAD * SEQ;
    _Float16* plds = plds_all[wv];

    const float scale = 0.125f;  // 1/sqrt(64)
    float mrow[8], lrow[8];
    v8f o0 = {}, o1 = {}, o2 = {}, o3 = {};
    for (int j = 0; j < 8; ++j) { mrow[j] = -1e30f; lrow[j] = 0.0f; }

    for (int t0 = 0; t0 < SEQ; t0 += 32) {
        // ---- scores: Q (16x64) * K^T (64x32) -> two 16x16 tiles ----
        v8f s0 = {}, s1 = {};
        {
            const _Float16* k0r = kbase + (long)(t0 + l16) * D_HEAD;        // keys t0..t0+15
            const _Float16* k1r = kbase + (long)(t0 + 16 + l16) * D_HEAD;   // keys t0+16..t0+31
            __builtin_prefetch(k0r + 32 * D_HEAD, 0, 3);
            s0 = wmma_f16(qf0, load_bfrag(k0r, lh), s0);
            s0 = wmma_f16(qf1, load_bfrag(k0r + 32, lh), s0);
            s1 = wmma_f16(qf0, load_bfrag(k1r, lh), s1);
            s1 = wmma_f16(qf1, load_bfrag(k1r + 32, lh), s1);
        }

        // ---- online softmax (rows spread across 16-lane groups) ----
        float p0[8], p1[8], corr[8];
        for (int j = 0; j < 8; ++j) {
            float a0 = s0[j] * scale;
            float a1 = s1[j] * scale;
            float mx = fmaxf(a0, a1);
            for (int msk = 1; msk < 16; msk <<= 1)
                mx = fmaxf(mx, __shfl_xor(mx, msk, 32));
            float mn = fmaxf(mrow[j], mx);
            float c  = __expf(mrow[j] - mn);
            mrow[j] = mn;
            corr[j] = c;
            p0[j] = __expf(a0 - mn);
            p1[j] = __expf(a1 - mn);
            float rs = p0[j] + p1[j];
            for (int msk = 1; msk < 16; msk <<= 1)
                rs += __shfl_xor(rs, msk, 32);
            lrow[j] = lrow[j] * c + rs;
        }
        for (int j = 0; j < 8; ++j) {
            o0[j] *= corr[j]; o1[j] *= corr[j]; o2[j] *= corr[j]; o3[j] *= corr[j];
        }

        // ---- C-layout -> A-fragment re-layout through LDS ----
        for (int j = 0; j < 8; ++j) {
            int row = j + 8 * lh;                     // query row within tile
            plds[row * 32 + l16]      = (_Float16)p0[j];
            plds[row * 32 + 16 + l16] = (_Float16)p1[j];
        }
        asm volatile("s_wait_dscnt 0" ::: "memory");
        v16h pf;
        {
            const _Float16* pr = plds + l16 * 32;     // A row = lane%16
            *((v8h*)&pf)     = *(const v8h*)(pr + 8 * lh);
            *((v8h*)&pf + 1) = *(const v8h*)(pr + 8 * lh + 16);
        }
        asm volatile("s_wait_dscnt 0" ::: "memory");

        // ---- ctx += P (16x32) * V (32x64) : 4 N-tiles, shared A ----
        o0 = wmma_f16(pf, load_bfrag(vbase + (long)(0  + l16) * SEQ + t0, lh), o0);
        o1 = wmma_f16(pf, load_bfrag(vbase + (long)(16 + l16) * SEQ + t0, lh), o1);
        o2 = wmma_f16(pf, load_bfrag(vbase + (long)(32 + l16) * SEQ + t0, lh), o2);
        o3 = wmma_f16(pf, load_bfrag(vbase + (long)(48 + l16) * SEQ + t0, lh), o3);
    }

    // ---- normalize and store ctx [B, S, H*64] as f16 ----
    _Float16* cbase = ctx + ((long)b * SEQ) * (HEADS * D_HEAD) + (long)h * D_HEAD;
    for (int j = 0; j < 8; ++j) {
        int row = m0 + j + 8 * lh;
        float inv = (lrow[j] > 0.0f) ? (1.0f / lrow[j]) : 0.0f;
        _Float16* cr = cbase + (long)row * (HEADS * D_HEAD);
        cr[l16]      = (_Float16)(o0[j] * inv);
        cr[16 + l16] = (_Float16)(o1[j] * inv);
        cr[32 + l16] = (_Float16)(o2[j] * inv);
        cr[48 + l16] = (_Float16)(o3[j] * inv);
    }
}

// ---------- output projection ----------
// out[B*S, 1024] = ctx[B*S, 1024] * Wo ; fp32 stores.
// 64x32 tile per wave (4Mx2N register blocking).  grid: 512 blocks of 128.
__global__ void out_proj(const _Float16* __restrict__ ctxh,
                         const _Float16* __restrict__ wo_t,
                         float* __restrict__ out) {
    const int lane = threadIdx.x & 31;
    const int wv   = threadIdx.x >> 5;
    const int lh   = lane >> 4;
    const int l16  = lane & 15;

    const int NB   = D_MODEL / 32;              // 32 N-blocks
    const int tile = blockIdx.x * 4 + wv;       // 64 M-blocks x 32 N-blocks = 2048
    const int m0   = (tile / NB) * 64;
    const int n0   = (tile % NB) * 32;

    const _Float16* ar0 = ctxh + (long)(m0 + l16) * D_MODEL;
    const _Float16* ar1 = ar0 + (long)16 * D_MODEL;
    const _Float16* ar2 = ar0 + (long)32 * D_MODEL;
    const _Float16* ar3 = ar0 + (long)48 * D_MODEL;
    const _Float16* br0 = wo_t + (long)(n0 + l16) * D_MODEL;
    const _Float16* br1 = br0 + (long)16 * D_MODEL;

    v8f c00 = {}, c01 = {}, c10 = {}, c11 = {};
    v8f c20 = {}, c21 = {}, c30 = {}, c31 = {};
    for (int k0 = 0; k0 < D_MODEL; k0 += 32) {
        __builtin_prefetch(ar0 + k0 + 32, 0, 3);
        __builtin_prefetch(br0 + k0 + 32, 0, 3);
        v16h b0 = load_bfrag(br0 + k0, lh);
        v16h b1 = load_bfrag(br1 + k0, lh);
        v16h a0 = load_afrag(ar0 + k0, lh);
        v16h a1 = load_afrag(ar1 + k0, lh);
        v16h a2 = load_afrag(ar2 + k0, lh);
        v16h a3 = load_afrag(ar3 + k0, lh);
        c00 = wmma_f16(a0, b0, c00);  c01 = wmma_f16(a0, b1, c01);
        c10 = wmma_f16(a1, b0, c10);  c11 = wmma_f16(a1, b1, c11);
        c20 = wmma_f16(a2, b0, c20);  c21 = wmma_f16(a2, b1, c21);
        c30 = wmma_f16(a3, b0, c30);  c31 = wmma_f16(a3, b1, c31);
    }
    for (int j = 0; j < 8; ++j) {
        long r = (long)(m0 + j + 8 * lh) * D_MODEL + n0 + l16;
        out[r]                          = c00[j];
        out[r + 16]                     = c01[j];
        out[r + (long)16 * D_MODEL]      = c10[j];
        out[r + (long)16 * D_MODEL + 16] = c11[j];
        out[r + (long)32 * D_MODEL]      = c20[j];
        out[r + (long)32 * D_MODEL + 16] = c21[j];
        out[r + (long)48 * D_MODEL]      = c30[j];
        out[r + (long)48 * D_MODEL + 16] = c31[j];
    }
}

// ---------- host launcher ----------
extern "C" void kernel_launch(void* const* d_in, const int* in_sizes, int n_in,
                              void* d_out, int out_size, void* d_ws, size_t ws_size,
                              hipStream_t stream) {
    (void)in_sizes; (void)n_in; (void)out_size; (void)ws_size;
    // setup_inputs order: x, Wk, Wq, Wv, Wo
    const float* x  = (const float*)d_in[0];
    const float* Wk = (const float*)d_in[1];
    const float* Wq = (const float*)d_in[2];
    const float* Wv = (const float*)d_in[3];
    const float* Wo = (const float*)d_in[4];
    float* out = (float*)d_out;

    char* ws = (char*)d_ws;
    size_t off = 0;
    auto alloc_h = [&](size_t halves) -> _Float16* {
        _Float16* p = (_Float16*)(ws + off);
        off += ((halves * sizeof(_Float16) + 255) / 256) * 256;
        return p;
    };
    _Float16* xh   = alloc_h((size_t)BATCH * SEQ * D_MODEL);          // 8 MB
    _Float16* wq_t = alloc_h((size_t)HEADS * D_HEAD * D_MODEL);       // 2 MB
    _Float16* wk_t = alloc_h((size_t)HEADS * D_HEAD * D_MODEL);
    _Float16* wv_t = alloc_h((size_t)HEADS * D_HEAD * D_MODEL);
    _Float16* wo_t = alloc_h((size_t)D_MODEL * D_MODEL);              // 2 MB
    _Float16* Qh   = alloc_h((size_t)BATCH * HEADS * SEQ * D_HEAD);   // 8 MB
    _Float16* Kh   = alloc_h((size_t)BATCH * HEADS * SEQ * D_HEAD);
    _Float16* Vt   = alloc_h((size_t)BATCH * HEADS * D_HEAD * SEQ);   // [b,h,d,S]
    _Float16* ctxh = alloc_h((size_t)BATCH * SEQ * (HEADS * D_HEAD)); // 8 MB

    // 1) precision conversion + weight transposes
    {
        int n = BATCH * SEQ * D_MODEL;
        cvt_f32_f16<<<(n + 255) / 256, 256, 0, stream>>>(x, xh, n);
        int nw = HEADS * D_MODEL * D_HEAD;
        cvt_transpose_w<<<(nw + 255) / 256, 256, 0, stream>>>(Wq, wq_t, HEADS, D_MODEL, D_HEAD);
        cvt_transpose_w<<<(nw + 255) / 256, 256, 0, stream>>>(Wk, wk_t, HEADS, D_MODEL, D_HEAD);
        cvt_transpose_w<<<(nw + 255) / 256, 256, 0, stream>>>(Wv, wv_t, HEADS, D_MODEL, D_HEAD);
        int no = D_MODEL * D_MODEL;
        cvt_transpose_w<<<(no + 255) / 256, 256, 0, stream>>>(Wo, wo_t, 1, D_MODEL, D_MODEL);
    }

    // 2) QKV projections (WMMA tile GEMMs, 4Mx2N register blocking)
    qkv_proj<<<dim3(16, BATCH * HEADS, 3), 128, 0, stream>>>(xh, wq_t, wk_t, wv_t, Qh, Kh, Vt);

    // 3) flash attention (WMMA scores + online softmax + WMMA PV)
    attn_kernel<<<dim3(SEQ / 64, BATCH * HEADS), 128, 0, stream>>>(Qh, Kh, Vt, ctxh);

    // 4) output projection (WMMA, fp32 out, 4Mx2N register blocking)
    out_proj<<<dim3((BATCH * SEQ / 64) * (D_MODEL / 32) / 4), 128, 0, stream>>>(ctxh, wo_t, out);
}
